// SpecialTransformerLayer_38233798869265
// MI455X (gfx1250) — matrix operand order repaired and checked by reading, not compiled, not run on hardware
//
#include <hip/hip_runtime.h>

// ---------------------------------------------------------------------------
// SpecialTransformerLayer for MI455X (gfx1250, wave32, WMMA).
// bf16 WMMA (v_wmma_f32_16x16x32_bf16) for all GEMMs + flash attention,
// fp32 softmax / groupnorm / gelu / residuals.
// GEMM waves register-blocked 16x64 (A-fragment reuse across 4 accumulators).
// ---------------------------------------------------------------------------

typedef __bf16 bf16;
typedef __attribute__((ext_vector_type(16))) __bf16 v16bf;
typedef __attribute__((ext_vector_type(8)))  __bf16 v8bf;
typedef __attribute__((ext_vector_type(8)))  float  v8f;

#define RSQRT2 0.70710678118654752440f

__device__ __forceinline__ v16bf join16(v8bf lo, v8bf hi) {
  return __builtin_shufflevector(lo, hi, 0,1,2,3,4,5,6,7,8,9,10,11,12,13,14,15);
}
__device__ __forceinline__ v8f wmma_bf16(v16bf a, v16bf b, v8f c) {
  return __builtin_amdgcn_wmma_f32_16x16x32_bf16(false, a, false, b, (short)0, c, false, false);
}

// --------------------------- weight transpose ------------------------------
// W (Cin x Cout fp32, row-major over o) -> WT (Cout x Cin bf16, row-major over c)
__global__ void k_packwt(const float* __restrict__ W, bf16* __restrict__ WT,
                         int Cin, int Cout) {
  int i = blockIdx.x * 256 + threadIdx.x;
  if (i < Cin * Cout) {
    int c = i / Cout, o = i - c * Cout;
    WT[(size_t)o * Cin + c] = (bf16)W[i];
  }
}

// ------------------------------ group norm ---------------------------------
// in: fp32 (B,128,P) channel-major.  out: bf16 position-major [(b*P+p)*128 + c].
// One block per (b, group); GROUPS=32, 4 channels per group.
__global__ void __launch_bounds__(256) k_groupnorm(
    const float* __restrict__ in, const float* __restrict__ gamma,
    const float* __restrict__ beta, bf16* __restrict__ outpm, int pshift) {
  const int C = 128;
  int P = 1 << pshift;
  int b = blockIdx.x >> 5;
  int g = blockIdx.x & 31;
  const float* src = in + ((size_t)b * C + g * 4) * P;
  int N = 4 * P;
  float s = 0.f, s2 = 0.f;
  for (int i = threadIdx.x; i < N; i += 256) {
    float v = src[i]; s += v; s2 += v * v;
  }
  __shared__ float r1[256], r2[256];
  r1[threadIdx.x] = s; r2[threadIdx.x] = s2;
  __syncthreads();
  for (int st = 128; st > 0; st >>= 1) {
    if (threadIdx.x < st) { r1[threadIdx.x] += r1[threadIdx.x + st];
                            r2[threadIdx.x] += r2[threadIdx.x + st]; }
    __syncthreads();
  }
  float mu  = r1[0] / (float)N;
  float var = r2[0] / (float)N - mu * mu;
  float inv = rsqrtf(var + 1e-6f);
  for (int i = threadIdx.x; i < N; i += 256) {
    int cc = i >> pshift;
    int p  = i & (P - 1);
    int c  = g * 4 + cc;
    float v = (src[i] - mu) * inv * gamma[c] + beta[c];
    outpm[((size_t)(b << pshift) + p) * C + c] = (bf16)v;
  }
}

// ------------------------------ linear (GEMM) ------------------------------
// Out[o][p] = sum_c WT[o][c] * X[p][c]  (+bias)*scale, epilogue per MODE:
//  MODE 0: bf16 position-major out[(p)*Cout + o]  (optional exact GELU)
//  MODE 1: bf16 channel-major  out[(o)*TotalP + p]
//  MODE 2: fp32 original layout out[((b*128+o)<<pshift)+p] = (skip+val)/sqrt2
// Each wave: one 16(o) x 64(p) tile = 4 accumulators sharing the A fragment.
template<int MODE, bool GELU>
__global__ void __launch_bounds__(256) k_linear(
    const bf16* __restrict__ X, const bf16* __restrict__ WT,
    const float* __restrict__ bias, float scale,
    bf16* __restrict__ outb, float* __restrict__ outf,
    const float* __restrict__ skip,
    int Cin, int Cout, int TotalP, int pshift) {
  int lane = threadIdx.x & 31;
  int wv   = blockIdx.x * 8 + (threadIdx.x >> 5);
  int otiles = Cout >> 4;
  int ot = wv % otiles;
  int pt = wv / otiles;
  int o0 = ot << 4, p0 = pt << 6;            // 64 positions per wave
  int n = lane & 15, h = lane >> 4;

  const bf16* aptr = WT + (size_t)(o0 + n) * Cin + 16 * h;  // A row = o0+n
  const bf16* bptr = X  + (size_t)(p0 + n) * Cin + 16 * h;  // B col = p0+n (+16*nt)
  v8f acc[4] = {};
  for (int k0 = 0; k0 < Cin; k0 += 32) {
    v8bf alo = *(const v8bf*)(aptr + k0);
    v8bf ahi = *(const v8bf*)(aptr + k0 + 8);
    v16bf a = join16(alo, ahi);
#pragma unroll
    for (int nt = 0; nt < 4; ++nt) {
      v16bf b = *(const v16bf*)(bptr + (size_t)nt * 16 * Cin + k0);
      acc[nt] = wmma_bf16(a, b, acc[nt]);
    }
  }
#pragma unroll
  for (int nt = 0; nt < 4; ++nt) {
    int pn = p0 + nt * 16 + n;               // this lane's output column
    if (MODE == 0) {
      v8bf ov;
#pragma unroll
      for (int r = 0; r < 8; ++r) {
        int o = o0 + 8 * h + r;
        float v = (acc[nt][r] + bias[o]) * scale;
        if (GELU) v = 0.5f * v * (1.0f + erff(v * RSQRT2));
        ov[r] = (bf16)v;
      }
      *(v8bf*)(outb + (size_t)pn * Cout + o0 + 8 * h) = ov;
    } else if (MODE == 1) {
#pragma unroll
      for (int r = 0; r < 8; ++r) {
        int o = o0 + 8 * h + r;
        outb[(size_t)o * TotalP + pn] = (bf16)((acc[nt][r] + bias[o]) * scale);
      }
    } else {
      int bb = pn >> pshift;
      int pp = pn & ((1 << pshift) - 1);
#pragma unroll
      for (int r = 0; r < 8; ++r) {
        int o = o0 + 8 * h + r;
        size_t idx = ((size_t)(bb * 128 + o) << pshift) + pp;
        float v = (acc[nt][r] + bias[o]) * scale;
        outf[idx] = (skip[idx] + v) * RSQRT2;
      }
    }
  }
}

// ----------------------------- flash attention -----------------------------
// DH = per-head dim (32 or 64), DIM = 2*DH. Qt/Kt position-major (rows=pos,
// DIM cols, q pre-scaled by 1/sqrt(DIM)); Vcm channel-major (DIM x TotalPk).
// One wave = 16 queries of one (batch, head). Computes S^T = K*Q via WMMA so
// exp'd probs in D-regs are directly the A-fragment of the P*V WMMA.
template<int DH>
__device__ __forceinline__ v8f st_tile(const bf16* kbase, const v16bf* qf) {
  v8f s = {};
#pragma unroll
  for (int t = 0; t < DH / 32; ++t) {
    v8bf alo = *(const v8bf*)(kbase + 32 * t);
    v8bf ahi = *(const v8bf*)(kbase + 32 * t + 8);
    s = wmma_bf16(join16(alo, ahi), qf[t], s);
  }
  return s;
}

template<int DH>
__global__ void __launch_bounds__(128) k_flash(
    const bf16* __restrict__ Qt, const bf16* __restrict__ Kt,
    const bf16* __restrict__ Vcm, bf16* __restrict__ Ht,
    int Pq, int Pk, int TotalPk) {
  constexpr int DIM = 2 * DH;
  constexpr int NQ  = DH / 32;   // K-chunks of 32 over d
  constexpr int ND  = DH / 16;   // d chunks of 16 for O
  int lane = threadIdx.x & 31;
  int wv   = blockIdx.x * 4 + (threadIdx.x >> 5);
  int qtiles = Pq >> 4;
  int qt   = wv % qtiles;
  int rest = wv / qtiles;
  int head = rest & 1;
  int b    = rest >> 1;
  int n = lane & 15, h = lane >> 4;
  int q0 = qt << 4;

  const bf16* qbase = Qt + (size_t)(b * Pq + q0 + n) * DIM + head * DH;
  v16bf qf[NQ];
#pragma unroll
  for (int t = 0; t < NQ; ++t) qf[t] = *(const v16bf*)(qbase + 32 * t + 16 * h);

  v8f Oacc[ND] = {};
  float m = -3.0e38f, l = 0.0f;

  for (int k0 = 0; k0 < Pk; k0 += 32) {
    const bf16* kb0 = Kt + (size_t)(b * Pk + k0 + n)      * DIM + head * DH + 16 * h;
    const bf16* kb1 = Kt + (size_t)(b * Pk + k0 + 16 + n) * DIM + head * DH + 16 * h;
    v8f st0 = st_tile<DH>(kb0, qf);   // S^T rows = keys k0..k0+15
    v8f st1 = st_tile<DH>(kb1, qf);   // S^T rows = keys k0+16..k0+31

    // online softmax over keys for query q = lane&15 (both halves symmetric)
    float mloc = st0[0];
#pragma unroll
    for (int r = 1; r < 8; ++r) mloc = fmaxf(mloc, st0[r]);
#pragma unroll
    for (int r = 0; r < 8; ++r) mloc = fmaxf(mloc, st1[r]);
    mloc = fmaxf(mloc, __shfl_xor(mloc, 16, 32));
    float mnew = fmaxf(m, mloc);
    float sc = __expf(m - mnew);
    float p0a[8], p1a[8];
    float ls = 0.f;
#pragma unroll
    for (int r = 0; r < 8; ++r) { p0a[r] = __expf(st0[r] - mnew); ls += p0a[r]; }
#pragma unroll
    for (int r = 0; r < 8; ++r) { p1a[r] = __expf(st1[r] - mnew); ls += p1a[r]; }
    ls += __shfl_xor(ls, 16, 32);
    l = l * sc + ls;
    m = mnew;

    // probs: D-regs are exactly the A-fragment layout for P(16x32) x V(32x16)
    v16bf pf;
#pragma unroll
    for (int j = 0; j < 8; ++j) { pf[j] = (bf16)p0a[j]; pf[8 + j] = (bf16)p1a[j]; }

    float rs[8];
#pragma unroll
    for (int r = 0; r < 8; ++r) rs[r] = __shfl(sc, 8 * h + r, 32);

#pragma unroll
    for (int dt = 0; dt < ND; ++dt) {
      const bf16* vp = Vcm + (size_t)(head * DH + dt * 16 + n) * TotalPk + b * Pk + k0 + 8 * h;
      v16bf vb = join16(*(const v8bf*)vp, *(const v8bf*)(vp + 16));
      v8f o = Oacc[dt];
#pragma unroll
      for (int r = 0; r < 8; ++r) o[r] *= rs[r];
      Oacc[dt] = wmma_bf16(pf, vb, o);
    }
  }

  float li[8];
#pragma unroll
  for (int r = 0; r < 8; ++r) li[r] = __shfl(l, 8 * h + r, 32);
#pragma unroll
  for (int dt = 0; dt < ND; ++dt)
#pragma unroll
    for (int r = 0; r < 8; ++r)
      Ht[(size_t)(b * Pq + q0 + 8 * h + r) * DIM + head * DH + dt * 16 + n] =
          (bf16)(Oacc[dt][r] / li[r]);
}

// ------------------------------- launchers ---------------------------------
static void packwt(hipStream_t s, const float* W, bf16* WT, int Cin, int Cout) {
  int nb = (Cin * Cout + 255) / 256;
  k_packwt<<<nb, 256, 0, s>>>(W, WT, Cin, Cout);
}
static void gnorm(hipStream_t s, const float* in, const float* g, const float* b,
                  bf16* out, int pshift) {
  k_groupnorm<<<256, 256, 0, s>>>(in, g, b, out, pshift);
}
static void lin_pm(hipStream_t s, const bf16* X, const bf16* WT, const float* bias,
                   float scale, bool gelu, bf16* out, int Cin, int Cout, int TotalP) {
  int nb = ((Cout >> 4) * (TotalP >> 6)) / 8;
  if (gelu) k_linear<0,true ><<<nb,256,0,s>>>(X,WT,bias,scale,out,nullptr,nullptr,Cin,Cout,TotalP,0);
  else      k_linear<0,false><<<nb,256,0,s>>>(X,WT,bias,scale,out,nullptr,nullptr,Cin,Cout,TotalP,0);
}
static void lin_cm(hipStream_t s, const bf16* X, const bf16* WT, const float* bias,
                   bf16* out, int Cin, int Cout, int TotalP) {
  int nb = ((Cout >> 4) * (TotalP >> 6)) / 8;
  k_linear<1,false><<<nb,256,0,s>>>(X,WT,bias,1.0f,out,nullptr,nullptr,Cin,Cout,TotalP,0);
}
static void lin_res(hipStream_t s, const bf16* X, const bf16* WT, const float* bias,
                    const float* skip, float* out, int Cin, int TotalP, int pshift) {
  int nb = ((128 >> 4) * (TotalP >> 6)) / 8;
  k_linear<2,false><<<nb,256,0,s>>>(X,WT,bias,1.0f,nullptr,out,skip,Cin,128,TotalP,pshift);
}

// ------------------------------ entry point --------------------------------
extern "C" void kernel_launch(void* const* d_in, const int* in_sizes, int n_in,
                              void* d_out, int out_size, void* d_ws, size_t ws_size,
                              hipStream_t stream) {
  (void)in_sizes; (void)n_in; (void)out_size; (void)ws_size;
  const int B = 8, Px = 4096, Pc = 1024;
  const int XP = B * Px;   // 32768 image positions
  const int CP = B * Pc;   // 8192 memory positions

  const float* Pm[48];
  for (int i = 0; i < 48; ++i) Pm[i] = (const float*)d_in[i];
  const float* x_in = Pm[0];
  const float* c_in = Pm[1];
  const int CA1 = 2, CA2 = 14;
  // per cross-attn block: +0 gn_a_g +1 gn_a_b +2 gn_b_g +3 gn_b_b
  //                       +4 Wq +5 bq +6 Wk +7 bk +8 Wv +9 bv +10 Wo +11 bo

  float* out_x = (float*)d_out;
  float* out_c = (float*)d_out + (size_t)XP * 128;

  // ---- workspace arena ----
  char* base = (char*)d_ws; size_t off = 0;
  auto alloc = [&](size_t bytes) -> void* {
    void* p = base + off; off += (bytes + 255) & ~(size_t)255; return p;
  };
  bf16* wtq1 = (bf16*)alloc(64*128*2);  bf16* wtk1 = (bf16*)alloc(64*128*2);
  bf16* wtv1 = (bf16*)alloc(64*128*2);  bf16* wto1 = (bf16*)alloc(128*64*2);
  bf16* wtq2 = (bf16*)alloc(64*128*2);  bf16* wtk2 = (bf16*)alloc(64*128*2);
  bf16* wtv2 = (bf16*)alloc(64*128*2);  bf16* wto2 = (bf16*)alloc(128*64*2);
  bf16* wtsq = (bf16*)alloc(128*128*2); bf16* wtsk = (bf16*)alloc(128*128*2);
  bf16* wtsv = (bf16*)alloc(128*128*2); bf16* wtso = (bf16*)alloc(128*128*2);
  bf16* wtac = (bf16*)alloc(256*128*2); bf16* wtbc = (bf16*)alloc(128*256*2);
  bf16* wta  = (bf16*)alloc(256*128*2); bf16* wtb  = (bf16*)alloc(128*256*2);
  bf16* bufA   = (bf16*)alloc((size_t)XP*128*2);  // GN(x) scratch (reused)
  bf16* bufB   = (bf16*)alloc((size_t)XP*64*2);   // K1t -> Q2t
  bf16* bufC   = (bf16*)alloc((size_t)XP*64*2);   // V1cm -> H2t
  bf16* bufMid = (bf16*)alloc((size_t)XP*256*2);  // FFN mid (x & cells)
  bf16* cn1  = (bf16*)alloc((size_t)CP*128*2);
  bf16* q1t  = (bf16*)alloc((size_t)CP*64*2);
  bf16* h1t  = (bf16*)alloc((size_t)CP*64*2);
  bf16* cnS  = (bf16*)alloc((size_t)CP*128*2);
  bf16* qS   = (bf16*)alloc((size_t)CP*128*2);
  bf16* kS   = (bf16*)alloc((size_t)CP*128*2);
  bf16* vS   = (bf16*)alloc((size_t)CP*128*2);
  bf16* hS   = (bf16*)alloc((size_t)CP*128*2);
  bf16* cnF  = (bf16*)alloc((size_t)CP*128*2);
  bf16* cn2  = (bf16*)alloc((size_t)CP*128*2);
  bf16* k2t  = (bf16*)alloc((size_t)CP*64*2);
  bf16* v2cm = (bf16*)alloc((size_t)CP*64*2);
  float* cells_a = (float*)alloc((size_t)CP*128*4);
  float* cells_b = (float*)alloc((size_t)CP*128*4);
  float* x_a     = (float*)alloc((size_t)XP*128*4);

  const float qs64  = 0.125f;               // 1/sqrt(64)
  const float qs128 = 0.08838834764831845f; // 1/sqrt(128)

  // ---- pack weights (transpose to [o][c], bf16) ----
  packwt(stream, Pm[CA1+4],  wtq1, 128, 64);
  packwt(stream, Pm[CA1+6],  wtk1, 128, 64);
  packwt(stream, Pm[CA1+8],  wtv1, 128, 64);
  packwt(stream, Pm[CA1+10], wto1, 64, 128);
  packwt(stream, Pm[CA2+4],  wtq2, 128, 64);
  packwt(stream, Pm[CA2+6],  wtk2, 128, 64);
  packwt(stream, Pm[CA2+8],  wtv2, 128, 64);
  packwt(stream, Pm[CA2+10], wto2, 64, 128);
  packwt(stream, Pm[28], wtsq, 128, 128);
  packwt(stream, Pm[29], wtsk, 128, 128);
  packwt(stream, Pm[30], wtsv, 128, 128);
  packwt(stream, Pm[31], wtso, 128, 128);
  packwt(stream, Pm[40], wtac, 128, 256);
  packwt(stream, Pm[42], wtbc, 256, 128);
  packwt(stream, Pm[44], wta,  128, 256);
  packwt(stream, Pm[46], wtb,  256, 128);

  // ---- ca1: pull x -> cells ----
  gnorm(stream, x_in, Pm[CA1+0], Pm[CA1+1], bufA, 12);       // GN(x)
  gnorm(stream, c_in, Pm[CA1+2], Pm[CA1+3], cn1, 10);        // GN(cells)
  lin_pm(stream, cn1,  wtq1, Pm[CA1+5], qs64, false, q1t, 128, 64, CP);
  lin_pm(stream, bufA, wtk1, Pm[CA1+7], 1.0f, false, bufB, 128, 64, XP);
  lin_cm(stream, bufA, wtv1, Pm[CA1+9], bufC, 128, 64, XP);
  k_flash<32><<<(B*2*(Pc>>4))/4, 128, 0, stream>>>(q1t, bufB, bufC, h1t, Pc, Px, XP);
  lin_res(stream, h1t, wto1, Pm[CA1+11], c_in, cells_a, 64, CP, 10);

  // ---- self attention over cells ----
  gnorm(stream, cells_a, Pm[26], Pm[27], cnS, 10);
  lin_pm(stream, cnS, wtsq, Pm[32], qs128, false, qS, 128, 128, CP);
  lin_pm(stream, cnS, wtsk, Pm[33], 1.0f,  false, kS, 128, 128, CP);
  lin_cm(stream, cnS, wtsv, Pm[34], vS, 128, 128, CP);
  k_flash<64><<<(B*2*(Pc>>4))/4, 128, 0, stream>>>(qS, kS, vS, hS, Pc, Pc, CP);
  lin_res(stream, hS, wtso, Pm[35], cells_a, cells_b, 128, CP, 10);

  // ---- FFN on cells (residual from ORIGINAL cells input) ----
  gnorm(stream, cells_b, Pm[36], Pm[37], cnF, 10);
  lin_pm(stream, cnF, wtac, Pm[41], 1.0f, true, bufMid, 128, 256, CP);
  lin_res(stream, bufMid, wtbc, Pm[43], c_in, out_c, 256, CP, 10);

  // ---- ca2: pull cells -> x ----
  gnorm(stream, out_c, Pm[CA2+0], Pm[CA2+1], cn2, 10);       // GN(cells_final)
  gnorm(stream, x_in,  Pm[CA2+2], Pm[CA2+3], bufA, 12);      // GN(x)
  lin_pm(stream, bufA, wtq2, Pm[CA2+5], qs64, false, bufB, 128, 64, XP);
  lin_pm(stream, cn2,  wtk2, Pm[CA2+7], 1.0f, false, k2t, 128, 64, CP);
  lin_cm(stream, cn2,  wtv2, Pm[CA2+9], v2cm, 128, 64, CP);
  k_flash<32><<<(B*2*(Px>>4))/4, 128, 0, stream>>>(bufB, k2t, v2cm, bufC, Px, Pc, CP);
  lin_res(stream, bufC, wto2, Pm[CA2+11], x_in, x_a, 64, XP, 12);

  // ---- FFN on x ----
  gnorm(stream, x_a, Pm[38], Pm[39], bufA, 12);
  lin_pm(stream, bufA, wta, Pm[45], 1.0f, true, bufMid, 128, 256, XP);
  lin_res(stream, bufMid, wtb, Pm[47], x_a, out_x, 256, XP, 12);
}
